// GraphTransformer_4836133175909
// MI455X (gfx1250) — matrix operand order repaired and checked by reading, not compiled, not run on hardware
//
#include <hip/hip_runtime.h>
#include <hip/hip_bf16.h>
#include <math.h>

// Problem constants (from the reference)
#define NNODES 50000
#define NEDGES 800000
#define FIN    128
#define HIDC   32
#define EMBC   64
#define NHEADS 4
#define NLABEL 128
#define EPS_   1e-16f

typedef __attribute__((ext_vector_type(16))) __bf16 v16bf;
typedef __attribute__((ext_vector_type(8)))  float  v8f;

// ---------- helpers ----------
__device__ __forceinline__ unsigned short f32_to_bf16_rne(float f) {
  unsigned int u = __float_as_uint(f);
  unsigned int r = 0x7FFFu + ((u >> 16) & 1u);
  return (unsigned short)((u + r) >> 16);
}
// order-preserving float <-> uint encoding for atomicMax on floats
__device__ __forceinline__ unsigned int enc_f(float f) {
  unsigned int u = __float_as_uint(f);
  return (u & 0x80000000u) ? ~u : (u | 0x80000000u);
}
__device__ __forceinline__ float dec_f(unsigned int u) {
  return __uint_as_float((u & 0x80000000u) ? (u & 0x7FFFFFFFu) : ~u);
}

// ---------- conversion kernels ----------
__global__ void k_cvt_bf16(const float* __restrict__ in,
                           unsigned short* __restrict__ out, size_t n) {
  size_t i = (size_t)blockIdx.x * blockDim.x + threadIdx.x;
  if (i < n) out[i] = f32_to_bf16_rne(in[i]);
}

// W[K x Nc] (f32 row-major) -> Wt[Nc x K] (bf16): B stored transposed so
// WMMA B-fragment K-pairs are contiguous 32-bit loads.
__global__ void k_cvt_bf16_T(const float* __restrict__ W,
                             unsigned short* __restrict__ Wt, int K, int Nc) {
  int i = blockIdx.x * blockDim.x + threadIdx.x;
  if (i < K * Nc) {
    int kk = i / Nc, nn = i % Nc;
    Wt[(size_t)nn * K + kk] = f32_to_bf16_rne(W[i]);
  }
}

// ---------- WMMA GEMM:  C[M x Nc] = A[M x K] * B[K x Nc] + bias ----------
// A: bf16 row-major (lda=K). Bt: bf16, B transposed (Nc x K, ldb=K).
// One wave computes a 16x64 C stripe: 4 accumulators share one A fragment per
// K-step, so each A load feeds 4 v_wmma_f32_16x16x32_bf16 (4 independent
// accumulation chains also avoid WMMA->WMMA RAW hazards).
// Fragment layouts per CDNA5 ISA 7.12.2:
//   A (16x32): lane&15 = M; lanes>=16 get K+8 in VGPR0-3 / K+24 in VGPR4-7.
//   B (32x16): lane&15 = N; same K order as A (lanes>=16 get K+8 / K+24).
//   C/D f32:   VGPR r -> M = r + ((lane&16)?8:0), N = lane&15.
__global__ void k_wmma_gemm_bias(const unsigned short* __restrict__ A,
                                 const unsigned short* __restrict__ Bt,
                                 const float* __restrict__ bias,
                                 float* __restrict__ C,
                                 int M, int K, int Nc) {
  int wave = (int)((blockIdx.x * blockDim.x + threadIdx.x) >> 5);
  int lane = threadIdx.x & 31;
  int tilesN = Nc >> 6;                  // 64-wide wave stripes
  int tilesM = M >> 4;
  if (wave >= tilesM * tilesN) return;   // wave-uniform: EXEC stays all-1s
  int tm = wave / tilesN, tn = wave % tilesN;
  int row = lane & 15;
  int hi  = (lane & 16) ? 1 : 0;

  const unsigned short* arow = A + (size_t)(tm * 16 + row) * K;
  const unsigned short* brow[4];
#pragma unroll
  for (int t = 0; t < 4; ++t)
    brow[t] = Bt + (size_t)(tn * 64 + t * 16 + row) * K;

  v8f acc[4] = {v8f{}, v8f{}, v8f{}, v8f{}};
  for (int kb = 0; kb < K; kb += 32) {
    union { v16bf v; unsigned int d[8]; } a;
#pragma unroll
    for (int r = 0; r < 8; ++r) {
      int ka = kb + ((r & 4) << 2) + (hi << 3) + 2 * (r & 3); // +16 if r>=4, +8 if hi
      a.d[r] = *(const unsigned int*)(arow + ka);
    }
#pragma unroll
    for (int t = 0; t < 4; ++t) {
      union { v16bf v; unsigned int d[8]; } b;
#pragma unroll
      for (int r = 0; r < 8; ++r) {
        int kB = kb + ((r & 4) << 2) + (hi << 3) + 2 * (r & 3); // +16 if r>=4, +8 if hi
        b.d[r] = *(const unsigned int*)(brow[t] + kB);
      }
      acc[t] = __builtin_amdgcn_wmma_f32_16x16x32_bf16(
          /*neg_a=*/false, a.v, /*neg_b=*/false, b.v,
          /*c_mod=*/(short)0, acc[t], /*reuse_a=*/false, /*reuse_b=*/false);
    }
  }

  int mb = tm * 16 + hi * 8;
#pragma unroll
  for (int t = 0; t < 4; ++t) {
    int cn = tn * 64 + t * 16 + row;
    float bv = bias[cn];
#pragma unroll
    for (int r = 0; r < 8; ++r)
      C[(size_t)(mb + r) * Nc + cn] = acc[t][r] + bv;
  }
}

// ---------- edge attention kernels ----------
__global__ void k_init_softmax(unsigned int* __restrict__ smax,
                               float* __restrict__ denom,
                               float* __restrict__ agg,
                               int nh, size_t nagg) {
  size_t i = (size_t)blockIdx.x * blockDim.x + threadIdx.x;
  if (i < (size_t)nh) { smax[i] = 0x007FFFFFu; /* enc(-inf) */ denom[i] = 0.f; }
  if (i < nagg) agg[i] = 0.f;
}

__global__ void k_edge_score(const float* __restrict__ q, const float* __restrict__ kf,
                             const long long* __restrict__ src, const long long* __restrict__ dst,
                             float* __restrict__ score, unsigned int* __restrict__ smax,
                             int C, float scale) {
  int idx = blockIdx.x * blockDim.x + threadIdx.x;
  if (idx >= NEDGES * NHEADS) return;
  int e = idx / NHEADS, h = idx % NHEADS;
  int s = (int)src[e], d = (int)dst[e];
  int HC = NHEADS * C;
  const float* qp = q  + (size_t)d * HC + h * C;
  const float* kp = kf + (size_t)s * HC + h * C;
  float acc = 0.f;
  for (int c = 0; c < C; c += 4) {
    float4 qv = *(const float4*)(qp + c);
    float4 kv = *(const float4*)(kp + c);
    acc += qv.x * kv.x + qv.y * kv.y + qv.z * kv.z + qv.w * kv.w;
  }
  acc *= scale;
  score[idx] = acc;
  atomicMax(&smax[d * NHEADS + h], enc_f(acc));
}

__global__ void k_edge_expsum(float* __restrict__ score,
                              const unsigned int* __restrict__ smax,
                              const long long* __restrict__ dst,
                              float* __restrict__ denom) {
  int idx = blockIdx.x * blockDim.x + threadIdx.x;
  if (idx >= NEDGES * NHEADS) return;
  int e = idx / NHEADS, h = idx % NHEADS;
  int d = (int)dst[e];
  float m  = dec_f(smax[d * NHEADS + h]);
  float ex = __expf(score[idx] - m);
  score[idx] = ex;                       // reuse score buffer for exp values
  atomicAdd(&denom[d * NHEADS + h], ex);
}

__global__ void k_edge_aggregate(const float* __restrict__ ex,
                                 const float* __restrict__ denom,
                                 const float* __restrict__ v,
                                 const long long* __restrict__ src,
                                 const long long* __restrict__ dst,
                                 float* __restrict__ agg, int C) {
  size_t idx = (size_t)blockIdx.x * blockDim.x + threadIdx.x;
  int HC = NHEADS * C;
  if (idx >= (size_t)NEDGES * HC) return;
  int e = (int)(idx / HC);
  int j = (int)(idx % HC);
  int h = j / C;
  int s = (int)src[e], d = (int)dst[e];
  float alpha = ex[e * NHEADS + h] / (denom[d * NHEADS + h] + EPS_);
  atomicAdd(&agg[(size_t)d * HC + j], alpha * v[(size_t)s * HC + j]);
}

// ---------- elementwise epilogues ----------
__global__ void k_add_elu_to_bf16(const float* __restrict__ agg,
                                  const float* __restrict__ skip,
                                  unsigned short* __restrict__ outbf, size_t n) {
  size_t i = (size_t)blockIdx.x * blockDim.x + threadIdx.x;
  if (i < n) {
    float x = agg[i] + skip[i];
    float y = (x > 0.f) ? x : (__expf(x) - 1.f);  // ELU, alpha = 1
    outbf[i] = f32_to_bf16_rne(y);
  }
}

__global__ void k_sigmoid(const float* __restrict__ in, float* __restrict__ out, size_t n) {
  size_t i = (size_t)blockIdx.x * blockDim.x + threadIdx.x;
  if (i < n) out[i] = 1.f / (1.f + __expf(-in[i]));
}

// ---------- host orchestration ----------
extern "C" void kernel_launch(void* const* d_in, const int* in_sizes, int n_in,
                              void* d_out, int out_size, void* d_ws, size_t ws_size,
                              hipStream_t stream) {
  (void)in_sizes; (void)n_in; (void)out_size; (void)ws_size;

  // pytree (sorted-dict) flatten order:
  // [0]=edge_index(int64 2xE), [1]=final.W, [2]=final.b,
  // layer l in {0,1,2}: base=3+8l: k.W,k.b,q.W,q.b,skip.W,skip.b,v.W,v.b
  // [27]=x
  const long long* ei   = (const long long*)d_in[0];
  const long long* srcI = ei;
  const long long* dstI = ei + NEDGES;
  const float* finalW = (const float*)d_in[1];
  const float* finalb = (const float*)d_in[2];
  const float* x      = (const float*)d_in[27];

  // workspace carve (256B aligned)
  char* p = (char*)d_ws;
  auto carve = [&](size_t bytes) -> void* {
    void* r = (void*)p;
    p += (bytes + 255) & ~(size_t)255;
    return r;
  };
  const size_t maxFeat = (size_t)NNODES * 256;
  float* qbuf    = (float*)carve(maxFeat * 4);
  float* kbuf    = (float*)carve(maxFeat * 4);
  float* vbuf    = (float*)carve(maxFeat * 4);
  float* skipbuf = (float*)carve(maxFeat * 4);
  float* aggbuf  = (float*)carve(maxFeat * 4);
  float* scorebuf        = (float*)carve((size_t)NEDGES * NHEADS * 4);
  unsigned int* smaxbuf  = (unsigned int*)carve((size_t)NNODES * NHEADS * 4);
  float* denombuf        = (float*)carve((size_t)NNODES * NHEADS * 4);
  unsigned short* hbfA   = (unsigned short*)carve(maxFeat * 2);
  unsigned short* hbfB   = (unsigned short*)carve(maxFeat * 2);
  unsigned short* wtbuf  = (unsigned short*)carve((size_t)256 * 256 * 2);
  float* logits = qbuf;  // reuse after last layer

  // x -> bf16
  {
    size_t n = (size_t)NNODES * FIN;
    k_cvt_bf16<<<(unsigned)((n + 255) / 256), 256, 0, stream>>>(x, hbfA, n);
  }

  unsigned short* hcur  = hbfA;
  unsigned short* hnext = hbfB;
  const int douts[3] = {HIDC, HIDC, EMBC};

  for (int l = 0; l < 3; ++l) {
    const int C  = douts[l];
    const int Nc = NHEADS * C;   // 128 / 128 / 256
    const int K  = 128;          // din is 128 for all three layers
    const int base = 3 + l * 8;
    const float* Wk = (const float*)d_in[base + 0];
    const float* bk = (const float*)d_in[base + 1];
    const float* Wq = (const float*)d_in[base + 2];
    const float* bq = (const float*)d_in[base + 3];
    const float* Ws = (const float*)d_in[base + 4];
    const float* bs = (const float*)d_in[base + 5];
    const float* Wv = (const float*)d_in[base + 6];
    const float* bv = (const float*)d_in[base + 7];

    auto gemm = [&](const float* W, const float* b, float* out) {
      int nW = K * Nc;
      k_cvt_bf16_T<<<(nW + 255) / 256, 256, 0, stream>>>(W, wtbuf, K, Nc);
      int waves  = (NNODES / 16) * (Nc / 64);   // 16x64 stripe per wave
      int blocks = (waves + 7) / 8;             // 8 waves (256 threads) per block
      k_wmma_gemm_bias<<<blocks, 256, 0, stream>>>(hcur, wtbuf, b, out,
                                                   NNODES, K, Nc);
    };
    gemm(Wq, bq, qbuf);
    gemm(Wk, bk, kbuf);
    gemm(Wv, bv, vbuf);
    gemm(Ws, bs, skipbuf);

    {
      size_t nagg = (size_t)NNODES * Nc;  // >= NNODES*NHEADS
      k_init_softmax<<<(unsigned)((nagg + 255) / 256), 256, 0, stream>>>(
          smaxbuf, denombuf, aggbuf, NNODES * NHEADS, nagg);
    }
    float scale = 1.f / sqrtf((float)C);
    {
      int n = NEDGES * NHEADS;
      k_edge_score<<<(n + 255) / 256, 256, 0, stream>>>(
          qbuf, kbuf, srcI, dstI, scorebuf, smaxbuf, C, scale);
      k_edge_expsum<<<(n + 255) / 256, 256, 0, stream>>>(
          scorebuf, smaxbuf, dstI, denombuf);
    }
    {
      size_t n = (size_t)NEDGES * Nc;
      k_edge_aggregate<<<(unsigned)((n + 255) / 256), 256, 0, stream>>>(
          scorebuf, denombuf, vbuf, srcI, dstI, aggbuf, C);
    }
    {
      size_t n = (size_t)NNODES * Nc;
      k_add_elu_to_bf16<<<(unsigned)((n + 255) / 256), 256, 0, stream>>>(
          aggbuf, skipbuf, hnext, n);
    }
    unsigned short* t = hcur; hcur = hnext; hnext = t;
  }

  // final: logits = h3(50000x256) @ finalW(256x128) + b ; sigmoid -> d_out
  {
    const int K = NHEADS * EMBC;  // 256
    const int Nc = NLABEL;        // 128
    int nW = K * Nc;
    k_cvt_bf16_T<<<(nW + 255) / 256, 256, 0, stream>>>(finalW, wtbuf, K, Nc);
    int waves = (NNODES / 16) * (Nc / 64);
    k_wmma_gemm_bias<<<(waves + 7) / 8, 256, 0, stream>>>(
        hcur, wtbuf, finalb, logits, NNODES, K, Nc);
    size_t n = (size_t)NNODES * NLABEL;
    k_sigmoid<<<(unsigned)((n + 255) / 256), 256, 0, stream>>>(
        logits, (float*)d_out, n);
  }
}
